// MultiHeadAttention_1571958030875
// MI455X (gfx1250) — matrix operand order repaired and checked by reading, not compiled, not run on hardware
//
#include <hip/hip_runtime.h>
#include <hip/hip_bf16.h>

// MI455X / gfx1250 fused causal MHA:
//   k1: qkv GEMM (WMMA f16->f32) + bias + head-major f16 scatter (Q pre-scaled 1/8)
//   k2: flash attention; S=QK^T and O=PV on WMMA; V tiles staged by the
//       Tensor Data Mover (tensor_load_to_lds, double-buffered, s_wait_tensorcnt)
//   k3: output projection GEMM (WMMA) + bias -> f32 out
// Fragment loads are contiguous 16B chunks in k = 16*hi + s order, same for A and B (no repack).

typedef __attribute__((ext_vector_type(16))) _Float16 v16h;
typedef __attribute__((ext_vector_type(8)))  _Float16 v8h;
typedef __attribute__((ext_vector_type(4)))  _Float16 v4h;
typedef __attribute__((ext_vector_type(8)))  float    v8f;
typedef __attribute__((ext_vector_type(4)))  unsigned int v4u;
typedef __attribute__((ext_vector_type(8)))  int      v8i;
typedef __attribute__((ext_vector_type(4)))  int      v4i;

static constexpr int Bb = 4, Tt = 2048, Cc = 768, Hh = 12, Dd = 64;
static constexpr int M1 = Bb * Tt;       // 8192
static constexpr int N1 = 3 * Cc;        // 2304
static constexpr int K1 = Cc;            // 768
static constexpr int LDA = 40;           // padded LDS leading dim (halves)

static __device__ __forceinline__ v8f wmma_f16(v16h a, v16h b, v8f c) {
  return __builtin_amdgcn_wmma_f32_16x16x32_f16(false, a, false, b, (short)0, c, false, false);
}

// A-fragment (16x32 f16): lane m=lane%16, hi=lane/16.
// halves[0..15] = mem[m*ld + 16*hi .. +15]
static __device__ __forceinline__ v16h load_a_frag(const _Float16* base, int ld, int lane) {
  const int m = lane & 15, hi = lane >> 4;
  const _Float16* p = base + m * ld + 16 * hi;
  const v8h lo = *(const v8h*)(p);
  const v8h hh = *(const v8h*)(p + 8);
  return __builtin_shufflevector(lo, hh, 0, 1, 2, 3, 4, 5, 6, 7, 8, 9, 10, 11, 12, 13, 14, 15);
}

// B-fragment (32x16 f16) from an n-major (transposed) tile: element [n][k],
// lane n=lane%16, k = 16*(lane/16) + 0..15 contiguous.
static __device__ __forceinline__ v16h load_bt_frag(const _Float16* base, int ld, int lane) {
  const int n = lane & 15, h = lane >> 4;
  const _Float16* p = base + n * ld + 16 * h;
  const v8h lo = *(const v8h*)(p);
  const v8h hh = *(const v8h*)(p + 8);
  return __builtin_shufflevector(lo, hh, 0, 1, 2, 3, 4, 5, 6, 7, 8, 9, 10, 11, 12, 13, 14, 15);
}

// ---------------------------------------------------------------- GEMM 1: qkv
__global__ __launch_bounds__(256) void gemm_qkv_kernel(
    const float* __restrict__ X, const float* __restrict__ W,
    const float* __restrict__ bias,
    _Float16* __restrict__ qh, _Float16* __restrict__ kh, _Float16* __restrict__ vh) {
  __shared__ _Float16 As[128 * LDA];          // [m][k]
  __shared__ _Float16 Bt[128 * LDA];          // [n][k] transposed
  const int tid = threadIdx.x, lane = tid & 31, wid = tid >> 5;
  const int bm0 = blockIdx.x * 128, bn0 = blockIdx.y * 128;
  const int wm0 = (wid & 3) * 32, wn0 = (wid >> 2) * 64;

  v8f acc[2][4];
#pragma unroll
  for (int tm = 0; tm < 2; ++tm)
#pragma unroll
    for (int tn = 0; tn < 4; ++tn) acc[tm][tn] = (v8f)0.0f;

  for (int k0 = 0; k0 < K1; k0 += 32) {
    __syncthreads();
#pragma unroll
    for (int i = 0; i < 4; ++i) {             // A tile 128x32 f32 -> f16 (8B stores)
      const int r = (tid >> 3) + i * 32;
      const int c = (tid & 7) * 4;
      const float4 fv = *(const float4*)(X + (size_t)(bm0 + r) * K1 + k0 + c);
      v4h hv = { (_Float16)fv.x, (_Float16)fv.y, (_Float16)fv.z, (_Float16)fv.w };
      *(v4h*)(&As[r * LDA + c]) = hv;
    }
#pragma unroll
    for (int i = 0; i < 4; ++i) {             // B tile 32x128 f32 -> f16, transpose to [n][k]
      const int kr = (tid >> 5) + i * 8;
      const int c = (tid & 31) * 4;
      const float4 fv = *(const float4*)(W + (size_t)(k0 + kr) * N1 + bn0 + c);
      Bt[(c + 0) * LDA + kr] = (_Float16)fv.x;
      Bt[(c + 1) * LDA + kr] = (_Float16)fv.y;
      Bt[(c + 2) * LDA + kr] = (_Float16)fv.z;
      Bt[(c + 3) * LDA + kr] = (_Float16)fv.w;
    }
    if (k0 + 32 < K1) {                       // prefetch next K tile
      __builtin_prefetch(X + (size_t)(bm0 + (tid >> 3)) * K1 + k0 + 32 + (tid & 7) * 4, 0, 1);
      __builtin_prefetch(W + (size_t)(k0 + 32 + (tid >> 5)) * N1 + bn0 + (tid & 31) * 4, 0, 1);
    }
    __syncthreads();

    v16h af[2], bf[4];
#pragma unroll
    for (int tm = 0; tm < 2; ++tm) af[tm] = load_a_frag(&As[(wm0 + tm * 16) * LDA], LDA, lane);
#pragma unroll
    for (int tn = 0; tn < 4; ++tn) bf[tn] = load_bt_frag(&Bt[(wn0 + tn * 16) * LDA], LDA, lane);
#pragma unroll
    for (int tm = 0; tm < 2; ++tm)
#pragma unroll
      for (int tn = 0; tn < 4; ++tn) acc[tm][tn] = wmma_f16(af[tm], bf[tn], acc[tm][tn]);
  }

  const int hi = lane >> 4, cn = lane & 15;
#pragma unroll
  for (int tm = 0; tm < 2; ++tm) {
#pragma unroll
    for (int tn = 0; tn < 4; ++tn) {
      const int j = bn0 + wn0 + tn * 16 + cn;
      const float bv = bias[j];
      const int sel = j / Cc;
      const int rem = j - sel * Cc;
      const int head = rem >> 6, d = rem & 63;
      _Float16* dst = (sel == 0) ? qh : ((sel == 1) ? kh : vh);
      const float sc = (sel == 0) ? 0.125f : 1.0f;     // 1/sqrt(64)
#pragma unroll
      for (int v = 0; v < 8; ++v) {
        const int bt = bm0 + wm0 + tm * 16 + v + 8 * hi;
        const int bb = bt >> 11, t = bt & (Tt - 1);
        dst[(((size_t)bb * Hh + head) * Tt + t) * Dd + d] =
            (_Float16)((acc[tm][tn][v] + bv) * sc);
      }
    }
  }
}

// ------------------------------------------------------------ flash attention
// 4 waves / block, one wave per 16-query tile. V tiles staged via TDM.
__global__ __launch_bounds__(128) void attn_kernel(
    const _Float16* __restrict__ qh, const _Float16* __restrict__ kh,
    const _Float16* __restrict__ vh, _Float16* __restrict__ yh) {
  __shared__ _Float16 vstage[4][2][32 * Dd];            // per-wave double buffer (32KB)
  __shared__ _Float16 plds[4][16 * 32];                 // per-wave P staging (4KB)
  const int lane = threadIdx.x & 31, wid = threadIdx.x >> 5;
  const int h = blockIdx.y, b = blockIdx.z;
  const int bh = b * Hh + h;
  const int q0 = blockIdx.x * 64 + wid * 16;

  const _Float16* qp = qh + (size_t)bh * Tt * Dd;
  const _Float16* kp = kh + (size_t)bh * Tt * Dd;
  const _Float16* vp = vh + (size_t)bh * Tt * Dd;

  const v16h qa0 = load_a_frag(qp + (size_t)q0 * Dd, Dd, lane);
  const v16h qa1 = load_a_frag(qp + (size_t)q0 * Dd + 32, Dd, lane);

  v8f o[4];
#pragma unroll
  for (int t = 0; t < 4; ++t) o[t] = (v8f)0.0f;
  float row_m[8], row_l[8];
#pragma unroll
  for (int v = 0; v < 8; ++v) { row_m[v] = -1e30f; row_l[v] = 0.0f; }

  const int hi = lane >> 4, cn = lane & 15;
  _Float16* pl = plds[wid];
  const int nkt = (q0 >> 5) + 1;                        // causal, 32 keys/step

  // TDM issue: 2D D# (ISA 8.3/8.4): count=1, type=2 ("image"), data_size=2B,
  // tile 64x32, tensor row len 64, row stride 64. 6-arg toolchain form.
  auto issue_v_tdm = [&](int kt, int buf) {
    const unsigned long long ga =
        (unsigned long long)(uintptr_t)(vp + (size_t)(kt * 32) * Dd);
    const unsigned lds_off = (unsigned)(uintptr_t)(&vstage[wid][buf][0]);
    v4u g0 = { 1u,                                       // count=1
               lds_off,                                  // lds_addr
               (unsigned)ga,                             // global_addr[31:0]
               (unsigned)((ga >> 32) & 0x01FFFFFFu) | (2u << 30) };  // ga[56:32]|type=2
    v8i g1 = { (int)(1u << 16),                          // data_size=1 (2 bytes)
               (int)(64u << 16),                         // tensor_dim0 = 64
               (int)(2048u << 16),                       // tensor_dim1 = 2048
               (int)(64u << 16),                         // tile_dim0 = 64
               32,                                       // tile_dim1 = 32
               64,                                       // tensor_dim0_stride = 64
               0, 0 };
    v4i gz4 = { 0, 0, 0, 0 };
    v8i gz8 = { 0, 0, 0, 0, 0, 0, 0, 0 };
    __builtin_amdgcn_tensor_load_to_lds(g0, g1, gz4, gz4, gz8, 0);
  };

  issue_v_tdm(0, 0);

  for (int kt = 0; kt < nkt; ++kt) {
    const int key0 = kt * 32;
    if (kt + 1 < nkt) issue_v_tdm(kt + 1, (kt + 1) & 1);

    // K^T fragments: contiguous 32B per lane
    const v16h kb00 = *(const v16h*)(kp + (size_t)(key0 + cn) * Dd + 0 + hi * 16);
    const v16h kb01 = *(const v16h*)(kp + (size_t)(key0 + cn) * Dd + 32 + hi * 16);
    const v16h kb10 = *(const v16h*)(kp + (size_t)(key0 + 16 + cn) * Dd + 0 + hi * 16);
    const v16h kb11 = *(const v16h*)(kp + (size_t)(key0 + 16 + cn) * Dd + 32 + hi * 16);

    v8f s0 = (v8f)0.0f, s1 = (v8f)0.0f;
    s0 = wmma_f16(qa0, kb00, s0);
    s0 = wmma_f16(qa1, kb01, s0);
    s1 = wmma_f16(qa0, kb10, s1);
    s1 = wmma_f16(qa1, kb11, s1);

    // online softmax over the 32-key slab (C layout: row=v+8*hi, col=cn)
#pragma unroll
    for (int v = 0; v < 8; ++v) {
      const int row = q0 + v + 8 * hi;
      float a = (key0 + cn <= row) ? s0[v] : -1e30f;
      float c = (key0 + 16 + cn <= row) ? s1[v] : -1e30f;
      float mx = fmaxf(a, c);
      mx = fmaxf(mx, __shfl_xor(mx, 1));
      mx = fmaxf(mx, __shfl_xor(mx, 2));
      mx = fmaxf(mx, __shfl_xor(mx, 4));
      mx = fmaxf(mx, __shfl_xor(mx, 8));
      const float mnew = fmaxf(row_m[v], mx);
      const float p0 = __expf(a - mnew);
      const float p1 = __expf(c - mnew);
      float rs = p0 + p1;
      rs += __shfl_xor(rs, 1);
      rs += __shfl_xor(rs, 2);
      rs += __shfl_xor(rs, 4);
      rs += __shfl_xor(rs, 8);
      const float corr = __expf(row_m[v] - mnew);
      row_m[v] = mnew;
      row_l[v] = row_l[v] * corr + rs;
      o[0][v] *= corr; o[1][v] *= corr; o[2][v] *= corr; o[3][v] *= corr;
      const int lr = v + 8 * hi;
      pl[lr * 32 + cn] = (_Float16)p0;
      pl[lr * 32 + cn + 16] = (_Float16)p1;
    }
    asm volatile("s_wait_dscnt 0x0" ::: "memory");      // P stores -> P frag reads

    // wait for this iteration's V tile (keep the prefetched one in flight)
    if (kt + 1 < nkt) __builtin_amdgcn_s_wait_tensorcnt(1);
    else              __builtin_amdgcn_s_wait_tensorcnt(0);

    const v16h pf = load_a_frag(pl, 32, lane);          // P as 16x32 A-matrix
    const _Float16* vst = vstage[wid][kt & 1];
#pragma unroll
    for (int t = 0; t < 4; ++t) {
      v16h vb;
#pragma unroll
      for (int v = 0; v < 8; ++v)
#pragma unroll
        for (int pos = 0; pos < 2; ++pos) {
          const int k = hi * 16 + 2 * v + pos;
          vb[2 * v + pos] = vst[k * Dd + t * 16 + cn];
        }
      o[t] = wmma_f16(pf, vb, o[t]);
    }
  }

  float inv[8];
#pragma unroll
  for (int v = 0; v < 8; ++v) inv[v] = 1.0f / row_l[v];
#pragma unroll
  for (int t = 0; t < 4; ++t)
#pragma unroll
    for (int v = 0; v < 8; ++v) {
      const int row = q0 + v + 8 * hi;
      yh[((size_t)(b * Tt + row)) * Cc + h * Dd + t * 16 + cn] =
          (_Float16)(o[t][v] * inv[v]);
    }
}

// ------------------------------------------------------------- GEMM 2: proj
__global__ __launch_bounds__(256) void gemm_proj_kernel(
    const _Float16* __restrict__ Y, const float* __restrict__ W,
    const float* __restrict__ bias, float* __restrict__ out) {
  __shared__ _Float16 As[128 * LDA];
  __shared__ _Float16 Bt[128 * LDA];
  const int tid = threadIdx.x, lane = tid & 31, wid = tid >> 5;
  const int bm0 = blockIdx.x * 128, bn0 = blockIdx.y * 128;
  const int wm0 = (wid & 3) * 32, wn0 = (wid >> 2) * 64;

  v8f acc[2][4];
#pragma unroll
  for (int tm = 0; tm < 2; ++tm)
#pragma unroll
    for (int tn = 0; tn < 4; ++tn) acc[tm][tn] = (v8f)0.0f;

  for (int k0 = 0; k0 < Cc; k0 += 32) {
    __syncthreads();
#pragma unroll
    for (int i = 0; i < 4; ++i) {             // A tile 128x32 f16 direct
      const int r = (tid >> 3) + i * 32;
      const int c = (tid & 7) * 4;
      *(v4h*)(&As[r * LDA + c]) = *(const v4h*)(Y + (size_t)(bm0 + r) * Cc + k0 + c);
    }
#pragma unroll
    for (int i = 0; i < 4; ++i) {             // B tile 32x128 f32 -> f16, transposed
      const int kr = (tid >> 5) + i * 8;
      const int c = (tid & 31) * 4;
      const float4 fv = *(const float4*)(W + (size_t)(k0 + kr) * Cc + bn0 + c);
      Bt[(c + 0) * LDA + kr] = (_Float16)fv.x;
      Bt[(c + 1) * LDA + kr] = (_Float16)fv.y;
      Bt[(c + 2) * LDA + kr] = (_Float16)fv.z;
      Bt[(c + 3) * LDA + kr] = (_Float16)fv.w;
    }
    if (k0 + 32 < Cc) {
      __builtin_prefetch(Y + (size_t)(bm0 + (tid >> 3)) * Cc + k0 + 32 + (tid & 7) * 4, 0, 1);
      __builtin_prefetch(W + (size_t)(k0 + 32 + (tid >> 5)) * Cc + bn0 + (tid & 31) * 4, 0, 1);
    }
    __syncthreads();

    v16h af[2], bf[4];
#pragma unroll
    for (int tm = 0; tm < 2; ++tm) af[tm] = load_a_frag(&As[(wm0 + tm * 16) * LDA], LDA, lane);
#pragma unroll
    for (int tn = 0; tn < 4; ++tn) bf[tn] = load_bt_frag(&Bt[(wn0 + tn * 16) * LDA], LDA, lane);
#pragma unroll
    for (int tm = 0; tm < 2; ++tm)
#pragma unroll
      for (int tn = 0; tn < 4; ++tn) acc[tm][tn] = wmma_f16(af[tm], bf[tn], acc[tm][tn]);
  }

  const int hi = lane >> 4, cn = lane & 15;
#pragma unroll
  for (int tm = 0; tm < 2; ++tm)
#pragma unroll
    for (int tn = 0; tn < 4; ++tn) {
      const int col = bn0 + wn0 + tn * 16 + cn;
      const float bv = bias[col];
#pragma unroll
      for (int v = 0; v < 8; ++v) {
        const int row = bm0 + wm0 + tm * 16 + v + 8 * hi;
        out[(size_t)row * Cc + col] = acc[tm][tn][v] + bv;
      }
    }
}

// ---------------------------------------------------------------- launcher
extern "C" void kernel_launch(void* const* d_in, const int* in_sizes, int n_in,
                              void* d_out, int out_size, void* d_ws, size_t ws_size,
                              hipStream_t stream) {
  (void)in_sizes; (void)n_in; (void)out_size; (void)ws_size;
  const float* x      = (const float*)d_in[0];
  const float* w_attn = (const float*)d_in[1];
  const float* b_attn = (const float*)d_in[2];
  const float* w_proj = (const float*)d_in[3];
  const float* b_proj = (const float*)d_in[4];
  float* out = (float*)d_out;

  char* ws = (char*)d_ws;
  const size_t headsz = (size_t)Bb * Hh * Tt * Dd * sizeof(_Float16); // 12.6 MB
  _Float16* qh = (_Float16*)(ws);
  _Float16* kh = (_Float16*)(ws + headsz);
  _Float16* vh = (_Float16*)(ws + 2 * headsz);
  _Float16* yh = (_Float16*)(ws + 3 * headsz);          // 48 MB total

  dim3 g1(M1 / 128, N1 / 128);                          // 64 x 18
  gemm_qkv_kernel<<<g1, 256, 0, stream>>>(x, w_attn, b_attn, qh, kh, vh);

  dim3 g2(Tt / 64, Hh, Bb);                             // 32 x 12 x 4
  attn_kernel<<<g2, 128, 0, stream>>>(qh, kh, vh, yh);

  dim3 g3(M1 / 128, Cc / 128);                          // 64 x 6
  gemm_proj_kernel<<<g3, 256, 0, stream>>>(yh, w_proj, b_proj, out);
}
